// Conv3DCapsuleLayer_18030272708929
// MI455X (gfx1250) — matrix operand order repaired and checked by reading, not compiled, not run on hardware
//
#include <hip/hip_runtime.h>

// ---------------------------------------------------------------------------
// Conv3D capsule layer on gfx1250: bf16 WMMA implicit-GEMM conv + wave32 routing
// ---------------------------------------------------------------------------
//  B=2, D=H=W=24 (voxels=13824), IC=8, IA=32, C=8, A=32, taps=27, K=864, N=256
//  fold: n = ic*2 + b  (16 GEMM batch rows); reference reshape reads it back as
//  votes[b'][ic'] = conv[n = b'*8 + ic'] -- routing indexes votes that way.
// ---------------------------------------------------------------------------

typedef __attribute__((ext_vector_type(16))) __bf16 v16bf;
typedef __attribute__((ext_vector_type(8)))  float  v8f;

union ABfrag { uint4 q[2]; v16bf v; };

static __device__ __forceinline__ unsigned short f32_to_bf16(float f) {
  unsigned int u = __float_as_uint(f);
  unsigned int r = u + 0x7FFFu + ((u >> 16) & 1u);   // round-to-nearest-even
  return (unsigned short)(r >> 16);
}

#define VOX   13824          // 24*24*24
#define NROW  34             // padded w-row: w = -1..24 valid, 26..33 zero

// ---------------------------------------------------------------------------
// 1) W [864][256] f32 -> bf16 B-fragments: [(tap*16+nt)*32+lane][16 halves]
//    contiguous K order (matches A frags): lane<16 holds K=0..15, lane>=16 holds K=16..31, N=lane%16
// ---------------------------------------------------------------------------
__global__ void convert_weights_kernel(const float* __restrict__ W,
                                       unsigned short* __restrict__ wfrag) {
  int t = blockIdx.x * 256 + threadIdx.x;          // 27*16*32 = 13824 lane-slots
  if (t >= 27 * 16 * 32) return;
  int lane = t & 31;
  int nt   = (t >> 5) & 15;
  int tap  = t >> 9;
  int hi   = lane >> 4;
  int nc   = lane & 15;
  unsigned short* dst = wfrag + (size_t)t * 16;
  #pragma unroll
  for (int e = 0; e < 16; ++e) {
    int K = tap * 32 + hi * 16 + e;
    int N = nt * 16 + nc;
    dst[e] = f32_to_bf16(W[(size_t)K * 256 + N]);
  }
}

// ---------------------------------------------------------------------------
// 2) input [b][d][h][w][ic][ia] f32 -> xb bf16 [n=ic*2+b][vox][ia]
// ---------------------------------------------------------------------------
__global__ void convert_input_kernel(const float* __restrict__ in,
                                     unsigned short* __restrict__ xb) {
  size_t t = (size_t)blockIdx.x * 256 + threadIdx.x;
  if (t >= (size_t)2 * VOX * 8 * 32) return;
  int ia = (int)(t & 31);
  size_t r = t >> 5;                 // (b*VOX + vox)*8 + ic
  int ic = (int)(r & 7);  r >>= 3;   // b*VOX + vox
  int vox = (int)(r % VOX);
  int b   = (int)(r / VOX);
  int n   = ic * 2 + b;
  xb[(((size_t)n * VOX + vox) << 5) + ia] = f32_to_bf16(in[t]);
}

// ---------------------------------------------------------------------------
// 3) implicit-GEMM conv: one WG per (n,d,h) row. 8 waves, each owns two N16
//    tiles and both M16 tiles (w 0..31 padded). K loop = 27 taps of K=32.
// ---------------------------------------------------------------------------
__global__ __launch_bounds__(256) void conv_votes_kernel(
    const unsigned short* __restrict__ xb,   // [16][VOX][32] bf16
    const uint4*          __restrict__ wfrag,
    float*                __restrict__ votes) // [16][VOX][256] f32
{
  __shared__ unsigned short lds_x[9 * NROW * 32];    // 19,584 B, 64B-aligned rows

  const int h = blockIdx.x, d = blockIdx.y, n = blockIdx.z;
  const int tid = threadIdx.x;

  // ---- stage 3x3 halo rows (zero-padded) into LDS ----
  const uint4* xq = (const uint4*)xb;                // 1 uint4 = 8 bf16
  for (int idx = tid; idx < 9 * NROW; idx += 256) {
    int r  = idx / NROW, wi = idx % NROW;
    int dd = r / 3,      dh = r % 3;
    int ds_ = d + dd - 1, hs = h + dh - 1, ws_ = wi - 1;
    uint4 z0 = make_uint4(0,0,0,0), z1 = z0, z2 = z0, z3 = z0;
    if (ds_ >= 0 && ds_ < 24 && hs >= 0 && hs < 24 && ws_ >= 0 && ws_ < 24) {
      size_t vb = ((size_t)n * VOX + (size_t)(ds_*576 + hs*24 + ws_)) * 4;
      z0 = xq[vb + 0]; z1 = xq[vb + 1]; z2 = xq[vb + 2]; z3 = xq[vb + 3];
    }
    uint4* lp = (uint4*)&lds_x[idx * 32];
    lp[0] = z0; lp[1] = z1; lp[2] = z2; lp[3] = z3;
  }
  __syncthreads();

  const int wave = tid >> 5, lane = tid & 31;
  const int m = lane & 15, hi = lane >> 4;
  const int nt0 = wave * 2, nt1 = wave * 2 + 1;

  v8f acc00 = {}, acc01 = {}, acc10 = {}, acc11 = {};

  #pragma unroll 3
  for (int tap = 0; tap < 27; ++tap) {
    const int dd = tap / 9, dh = (tap / 3) % 3, dw = tap % 3;
    const int r = dd * 3 + dh;
    // A frags (contiguous K = hi*16..+15 for row M=m, matching wfrag)
    ABfrag a0, a1, b0, b1;
    {
      const uint4* ap = (const uint4*)&lds_x[(r*NROW + (m + dw)) * 32 + hi*16];
      a0.q[0] = ap[0]; a0.q[1] = ap[1];
      const uint4* ap1 = (const uint4*)&lds_x[(r*NROW + (m + 16 + dw)) * 32 + hi*16];
      a1.q[0] = ap1[0]; a1.q[1] = ap1[1];
    }
    {
      size_t i0 = ((size_t)(tap * 16 + nt0) * 32 + lane) * 2;
      size_t i1 = ((size_t)(tap * 16 + nt1) * 32 + lane) * 2;
      b0.q[0] = wfrag[i0]; b0.q[1] = wfrag[i0 + 1];
      b1.q[0] = wfrag[i1]; b1.q[1] = wfrag[i1 + 1];
    }
    acc00 = __builtin_amdgcn_wmma_f32_16x16x32_bf16(false, a0.v, false, b0.v,
                                                    (short)0, acc00, false, false);
    acc01 = __builtin_amdgcn_wmma_f32_16x16x32_bf16(false, a0.v, false, b1.v,
                                                    (short)0, acc01, false, false);
    acc10 = __builtin_amdgcn_wmma_f32_16x16x32_bf16(false, a1.v, false, b0.v,
                                                    (short)0, acc10, false, false);
    acc11 = __builtin_amdgcn_wmma_f32_16x16x32_bf16(false, a1.v, false, b1.v,
                                                    (short)0, acc11, false, false);
  }

  // ---- epilogue: C layout VGPR v, lane -> M=v+8*hi, N=lane%16 ----
  const size_t rowbase = ((size_t)n * VOX + (size_t)(d*576 + h*24)) * 256;
  const int ncol = lane & 15;
  #pragma unroll
  for (int v = 0; v < 8; ++v) {
    int M = v + hi * 8;
    if (M < 24) {                        // mt = 0 rows (w = M)
      votes[rowbase + (size_t)M*256 + nt0*16 + ncol] = acc00[v];
      votes[rowbase + (size_t)M*256 + nt1*16 + ncol] = acc01[v];
    }
    int M1 = 16 + M;                     // mt = 1 rows (w = 16+M), w>=24 padded
    if (M1 < 24) {
      votes[rowbase + (size_t)M1*256 + nt0*16 + ncol] = acc10[v];
      votes[rowbase + (size_t)M1*256 + nt1*16 + ncol] = acc11[v];
    }
  }
}

// ---------------------------------------------------------------------------
// 4) dynamic routing: WG = (voxel, b). wave = capsule c, lane = atom a.
//    votes live in registers across all 3 iterations.
// ---------------------------------------------------------------------------
__global__ __launch_bounds__(256) void routing_kernel(
    const float* __restrict__ votes, const float* __restrict__ bias,
    float* __restrict__ out)
{
  const int vox = blockIdx.x, b = blockIdx.y;
  const int tid = threadIdx.x;
  const int c = tid >> 5, a = tid & 31;

  __shared__ float logits[8][8];   // [ic][c]
  __shared__ float route[8][8];

  float v[8];
  #pragma unroll
  for (int ic = 0; ic < 8; ++ic) {
    size_t n = (size_t)(b * 8 + ic);                 // reference reshape order
    v[ic] = votes[(n * VOX + vox) * 256 + c * 32 + a];
  }
  const float bia = bias[c * 32 + a];

  if (tid < 64) logits[tid >> 3][tid & 7] = 0.f;
  __syncthreads();

  float act = 0.f;
  for (int it = 0; it < 3; ++it) {
    if (tid < 8) {                                   // softmax over c per ic
      int ic = tid;
      float mx = logits[ic][0];
      #pragma unroll
      for (int j = 1; j < 8; ++j) mx = fmaxf(mx, logits[ic][j]);
      float e[8], s = 0.f;
      #pragma unroll
      for (int j = 0; j < 8; ++j) { e[j] = __expf(logits[ic][j] - mx); s += e[j]; }
      float inv = 1.f / s;
      #pragma unroll
      for (int j = 0; j < 8; ++j) route[ic][j] = e[j] * inv;
    }
    __syncthreads();

    float pre = bia;
    #pragma unroll
    for (int ic = 0; ic < 8; ++ic) pre += route[ic][c] * v[ic];

    float n2 = pre * pre;                            // ||x||^2 over atoms
    #pragma unroll
    for (int off = 16; off >= 1; off >>= 1) n2 += __shfl_xor(n2, off, 32);
    float norm = sqrtf(n2);
    act = pre * (norm / (1.f + n2));                 // == x/||x|| * n2/(1+n2)

    if (it < 2) {
      __syncthreads();
      #pragma unroll
      for (int ic = 0; ic < 8; ++ic) {
        float dd = v[ic] * act;
        #pragma unroll
        for (int off = 16; off >= 1; off >>= 1) dd += __shfl_xor(dd, off, 32);
        if (a == 0) logits[ic][c] += dd;
      }
      __syncthreads();
    }
  }
  out[((size_t)(b * VOX + vox)) * 256 + c * 32 + a] = act;
}

// ---------------------------------------------------------------------------
extern "C" void kernel_launch(void* const* d_in, const int* in_sizes, int n_in,
                              void* d_out, int out_size, void* d_ws, size_t ws_size,
                              hipStream_t stream) {
  (void)in_sizes; (void)n_in; (void)out_size; (void)ws_size;
  const float* input = (const float*)d_in[0];   // [2,24,24,24,8,32]
  const float* W     = (const float*)d_in[1];   // [3,3,3,32,256]
  const float* bias  = (const float*)d_in[2];   // [8,32]
  float* out = (float*)d_out;

  char* ws = (char*)d_ws;
  const size_t WB_BYTES = (size_t)27 * 16 * 32 * 16 * 2;       //    442,368
  const size_t XB_BYTES = (size_t)16 * VOX * 32 * 2;           // 14,155,776
  unsigned short* wfrag = (unsigned short*)ws;
  unsigned short* xb    = (unsigned short*)(ws + WB_BYTES);
  float*          votes = (float*)(ws + WB_BYTES + XB_BYTES);  // 226,492,416 B

  convert_weights_kernel<<<(27*16*32 + 255)/256, 256, 0, stream>>>(W, wfrag);
  convert_input_kernel<<<(2*VOX*8*32 + 255)/256, 256, 0, stream>>>(input, xb);
  conv_votes_kernel<<<dim3(24, 24, 16), 256, 0, stream>>>(xb, (const uint4*)wfrag, votes);
  routing_kernel<<<dim3(VOX, 2), 256, 0, stream>>>(votes, bias, out);
}
